// TacticExpert_37529424233345
// MI455X (gfx1250) — matrix operand order repaired and checked by reading, not compiled, not run on hardware
//
#include <hip/hip_runtime.h>
#include <hip/hip_bf16.h>
#include <math.h>

typedef __attribute__((ext_vector_type(16))) _Float16 v16h;
typedef __attribute__((ext_vector_type(8)))  _Float16 v8h;
typedef __attribute__((ext_vector_type(8)))  float    v8f;

#define NE      5
#define HH      384
#define H2      192
#define OO      256
#define TOK     150      // T*P tokens per batch row
#define RDIM    150      // router input dim
#define KSTEPS  12       // 384 / 32
#define NTILES  24       // 384 / 16
#define MTILES  10       // ceil(150/16)
#define H1STR   392      // h1 A-tile LDS row stride (halves): 784B, 16B-aligned, bank-staggered
#define XSTR    40       // x  A-tile LDS row stride (halves): 80B, 16B-aligned, bank-staggered

// ---------------------------------------------------------------------------
// Prep A: We2[e] (f32 [K=384][N=384]) -> WMMA B-fragment layout (f16).
// ISA 7.12.2 (16-bit B 32x16): lane l -> N = nt*16+(l&15),
// halves i=0..7 -> K = ks*32 + (l>>4)*8 + i, i=8..15 -> K = ks*32 + 16 + (l>>4)*8 + (i-8).  One 32B fragment per lane.
// ---------------------------------------------------------------------------
__global__ void prep_bfrag2(const float* __restrict__ We2, _Float16* __restrict__ Bopt) {
    int id   = blockIdx.x;                  // e*288 + ks*24 + nt
    int nt   = id % NTILES;
    int ks   = (id / NTILES) % KSTEPS;
    int e    = id / (NTILES * KSTEPS);
    int lane = threadIdx.x;                 // 0..31
    int n    = nt * 16 + (lane & 15);
    int half = lane >> 4;
    _Float16* dst = Bopt + ((((size_t)e * KSTEPS + ks) * NTILES + nt) * 32 + lane) * 16;
    const float* src = We2 + (size_t)e * HH * HH;
    #pragma unroll
    for (int i = 0; i < 16; ++i) {
        int k = ks * 32 + (i < 8 ? half * 8 + i : 16 + half * 8 + (i - 8));
        dst[i] = (_Float16)src[(size_t)k * HH + n];
    }
}

// ---------------------------------------------------------------------------
// Prep B: We1[e] (f32 [K=10][N=384]) -> B-fragments with K zero-padded to 32.
// ---------------------------------------------------------------------------
__global__ void prep_bfrag1(const float* __restrict__ We1, _Float16* __restrict__ Bopt1) {
    int id   = blockIdx.x;                  // e*24 + nt
    int nt   = id % NTILES;
    int e    = id / NTILES;
    int lane = threadIdx.x;
    int n    = nt * 16 + (lane & 15);
    int half = lane >> 4;
    _Float16* dst = Bopt1 + (((size_t)e * NTILES + nt) * 32 + lane) * 16;
    #pragma unroll
    for (int i = 0; i < 16; ++i) {
        int k = (i < 8 ? half * 8 + i : 16 + half * 8 + (i - 8));
        dst[i] = (k < 10) ? (_Float16)We1[((size_t)e * 10 + k) * HH + n] : (_Float16)0.f;
    }
}

// ---------------------------------------------------------------------------
// Router: per batch row.  Two LN+ReLU MLP layers, logits, gumbel-argmax.
// routing_weights == onehot exactly in forward (y_soft - stopgrad(y_soft)==0).
// ---------------------------------------------------------------------------
__device__ __forceinline__ float reduce_r(float v, float* red, int tid) {
    red[tid] = v;
    __syncthreads();
    #pragma unroll
    for (int s = 256; s > 0; s >>= 1) {
        if (tid < s) red[tid] += red[tid + s];
        __syncthreads();
    }
    float r = red[0];
    __syncthreads();
    return r;
}

__global__ __launch_bounds__(384)
void router_kernel(const float* __restrict__ x,  const float* __restrict__ u,
                   const float* __restrict__ Wr1, const float* __restrict__ br1,
                   const float* __restrict__ lng1, const float* __restrict__ lnb1,
                   const float* __restrict__ Wr2, const float* __restrict__ br2,
                   const float* __restrict__ lng2, const float* __restrict__ lnb2,
                   const float* __restrict__ Wr3, const float* __restrict__ br3,
                   float* __restrict__ rw_out, float* __restrict__ idx_out_f,
                   int* __restrict__ idx_out_i) {
    __shared__ float rf[160];
    __shared__ float h1s[HH];
    __shared__ float h2s[H2];
    __shared__ float red[512];
    __shared__ float lg[8];
    __shared__ int   bsel;

    int b = blockIdx.x, tid = threadIdx.x;
    if (tid < 128) red[384 + tid] = 0.f;                 // pad reduction tail once
    if (tid < RDIM) rf[tid] = x[(size_t)b * 1500 + (tid / 10) * 100 + (tid % 10)];
    __syncthreads();

    // layer 1: 150 -> 384, LN, ReLU  (weight reads coalesced across threads)
    float h = br1[tid];
    for (int i = 0; i < RDIM; ++i) h = fmaf(rf[i], Wr1[(size_t)i * HH + tid], h);
    float sum = reduce_r(h, red, tid);
    float sq  = reduce_r(h * h, red, tid);
    float mu = sum * (1.f / 384.f);
    float var = sq * (1.f / 384.f) - mu * mu;
    h1s[tid] = fmaxf((h - mu) * rsqrtf(var + 1e-5f) * lng1[tid] + lnb1[tid], 0.f);
    __syncthreads();

    // layer 2: 384 -> 192, LN, ReLU
    float h2v = 0.f;
    if (tid < H2) {
        h2v = br2[tid];
        for (int i = 0; i < HH; ++i) h2v = fmaf(h1s[i], Wr2[(size_t)i * H2 + tid], h2v);
    }
    sum = reduce_r(tid < H2 ? h2v : 0.f, red, tid);
    sq  = reduce_r(tid < H2 ? h2v * h2v : 0.f, red, tid);
    mu  = sum * (1.f / 192.f);
    var = sq * (1.f / 192.f) - mu * mu;
    if (tid < H2)
        h2s[tid] = fmaxf((h2v - mu) * rsqrtf(var + 1e-5f) * lng2[tid] + lnb2[tid], 0.f);
    __syncthreads();

    // logits: 192 -> 5
    if (tid < NE) {
        float a = br3[tid];
        for (int i = 0; i < H2; ++i) a = fmaf(h2s[i], Wr3[(size_t)i * NE + tid], a);
        lg[tid] = a;
    }
    __syncthreads();

    // gumbel + argmax (tau > 0 preserves argmax; softmax never needed)
    if (tid == 0) {
        int bi = 0; float best = -1e30f;
        #pragma unroll
        for (int e = 0; e < NE; ++e) {
            float ue = u[(size_t)b * NE + e];
            float g  = -logf(-logf(ue + 1e-10f) + 1e-10f);
            float z  = lg[e] + g;
            if (z > best) { best = z; bi = e; }
        }
        bsel = bi;
        idx_out_i[b] = bi;
        idx_out_f[b] = (float)bi;
    }
    __syncthreads();
    if (tid < NE) rw_out[(size_t)b * NE + tid] = (tid == bsel) ? 1.f : 0.f;
}

// ---------------------------------------------------------------------------
// Expert + output projection: one block per batch row (8 wave32 waves).
// Only the argmax-selected expert is computed (onehot selection).
// Both MLP layers run on the WMMA pipe; B fragments are loop-invariant and
// get hoisted into VGPRs by the compiler (s_set_vgpr_msb high-VGPR path).
// ---------------------------------------------------------------------------
__device__ __forceinline__ float reduce_e(float v, float* red, int tid) {
    red[tid] = v;
    __syncthreads();
    #pragma unroll
    for (int s = 128; s > 0; s >>= 1) {
        if (tid < s) red[tid] += red[tid + s];
        __syncthreads();
    }
    float r = red[0];
    __syncthreads();
    return r;
}

__device__ __forceinline__ v16h load_afrag(const _Float16* hp) {
    v8h alo = *(const v8h*)hp;          // halves 0..7 : K = 8*half + i
    v8h ahi = *(const v8h*)(hp + 16);   // halves 8..15: K = 16 + 8*half + i
    return __builtin_shufflevector(alo, ahi,
        0, 1, 2, 3, 4, 5, 6, 7, 8, 9, 10, 11, 12, 13, 14, 15);
}

__global__ __launch_bounds__(256)
void expert_kernel(const float* __restrict__ x, const int* __restrict__ idx,
                   const float* __restrict__ be1, const float* __restrict__ be2,
                   const _Float16* __restrict__ Bopt1, const _Float16* __restrict__ Bopt2,
                   const float* __restrict__ Wo1, const float* __restrict__ bo1,
                   const float* __restrict__ lngo, const float* __restrict__ lnbo,
                   const float* __restrict__ Wo2, const float* __restrict__ bo2,
                   float* __restrict__ out) {
    __shared__ __align__(16) _Float16 h1t[16 * H1STR];   // 12.25 KB h2-input A-tile
    __shared__ __align__(16) _Float16 xA[16 * XSTR];     // 1.25 KB zero-padded x A-tile
    __shared__ float be1s[HH];
    __shared__ float be2s[HH];
    __shared__ float selacc[HH];
    __shared__ float red[256];
    __shared__ float hos[H2];

    int b = blockIdx.x, tid = threadIdx.x;
    int e = idx[b];

    for (int i = tid; i < HH; i += 256) {
        be1s[i] = be1[(size_t)e * HH + i];
        be2s[i] = be2[(size_t)e * HH + i];
        selacc[i] = 0.f;
    }

    int wave = tid >> 5, lane = tid & 31;
    int rown = lane & 15, half = lane >> 4;
    const _Float16* b1base = Bopt1 + (size_t)e * NTILES * 32 * 16;
    const _Float16* b2base = Bopt2 + (size_t)e * KSTEPS * NTILES * 32 * 16;

    for (int mt = 0; mt < MTILES; ++mt) {
        int tb = mt * 16;
        __syncthreads();                      // previous tile fully consumed

        // stage x tile as zero-padded f16 A-tile (16 rows x K=32, 10 valid K)
        for (int o = tid; o < 16 * 32; o += 256) {
            int t = o >> 5, k = o & 31;
            int tok = tb + t;
            float v = (k < 10 && tok < TOK) ? x[(size_t)b * 1500 + tok * 10 + k] : 0.f;
            xA[t * XSTR + k] = (_Float16)v;
        }
        __syncthreads();

        // ---- layer 1 on WMMA: h1 = relu(x @ We1 + be1), single K-step ----
        {
            v16h a = load_afrag(&xA[rown * XSTR + half * 8]);
            v8f d0 = {}; v8f d1 = {}; v8f d2 = {};
            v16h q0 = *(const v16h*)(b1base + ((size_t)(wave     ) * 32 + lane) * 16);
            v16h q1 = *(const v16h*)(b1base + ((size_t)(wave +  8) * 32 + lane) * 16);
            v16h q2 = *(const v16h*)(b1base + ((size_t)(wave + 16) * 32 + lane) * 16);
            d0 = __builtin_amdgcn_wmma_f32_16x16x32_f16(false, a, false, q0, (short)0, d0, false, false);
            d1 = __builtin_amdgcn_wmma_f32_16x16x32_f16(false, a, false, q1, (short)0, d1, false, false);
            d2 = __builtin_amdgcn_wmma_f32_16x16x32_f16(false, a, false, q2, (short)0, d2, false, false);
            v8f dd[3] = {d0, d1, d2};
            // epilogue: +bias, ReLU, f16, scatter into h2 A-tile.
            // Pad-token rows become relu(be1) but are masked out of the mean below.
            #pragma unroll
            for (int j = 0; j < 3; ++j) {
                int nt   = wave + j * 8;
                int ncol = nt * 16 + rown;            // C layout: N = lane&15
                float bias = be1s[ncol];
                #pragma unroll
                for (int i = 0; i < 8; ++i) {         // C layout: M = i + 8*half
                    int row = i + 8 * half;
                    h1t[row * H1STR + ncol] = (_Float16)fmaxf(dd[j][i] + bias, 0.f);
                }
            }
        }
        __syncthreads();

        // ---- layer 2 on WMMA: h2 = relu(h1 @ We2 + be2), 12 K-steps ----
        v8f c0 = {}; v8f c1 = {}; v8f c2 = {};
        for (int ks = 0; ks < KSTEPS; ++ks) {
            v16h a = load_afrag(&h1t[rown * H1STR + ks * 32 + half * 8]);
            const _Float16* kb = b2base + (size_t)ks * NTILES * 32 * 16;
            v16h p0 = *(const v16h*)(kb + ((size_t)(wave     ) * 32 + lane) * 16);
            v16h p1 = *(const v16h*)(kb + ((size_t)(wave +  8) * 32 + lane) * 16);
            v16h p2 = *(const v16h*)(kb + ((size_t)(wave + 16) * 32 + lane) * 16);
            c0 = __builtin_amdgcn_wmma_f32_16x16x32_f16(false, a, false, p0, (short)0, c0, false, false);
            c1 = __builtin_amdgcn_wmma_f32_16x16x32_f16(false, a, false, p1, (short)0, c1, false, false);
            c2 = __builtin_amdgcn_wmma_f32_16x16x32_f16(false, a, false, p2, (short)0, c2, false, false);
        }

        // epilogue: bias + ReLU + mask pad tokens + partial token-sum
        v8f cc[3] = {c0, c1, c2};
        #pragma unroll
        for (int j = 0; j < 3; ++j) {
            int nt   = wave + j * 8;
            int ncol = nt * 16 + rown;
            float bias = be2s[ncol];
            float s = 0.f;
            #pragma unroll
            for (int i = 0; i < 8; ++i) {
                int grow = tb + i + 8 * half;
                float v = fmaxf(cc[j][i] + bias, 0.f);
                if (grow < TOK) s += v;
            }
            s += __shfl_xor(s, 16);           // pair (lane, lane^16): same column
            if (half == 0) selacc[ncol] += s; // exclusive per-wave ownership of nt
        }
    }
    __syncthreads();

    // --- fused output projection: relu(LN(sel @ Wo1 + bo1)) @ Wo2 + bo2 ---
    float hv = 0.f;
    if (tid < H2) {
        hv = bo1[tid];
        for (int i = 0; i < HH; ++i)
            hv = fmaf(selacc[i] * (1.f / 150.f), Wo1[(size_t)i * H2 + tid], hv);
    }
    float sum = reduce_e(tid < H2 ? hv : 0.f, red, tid);
    float sq  = reduce_e(tid < H2 ? hv * hv : 0.f, red, tid);
    float mu  = sum * (1.f / 192.f);
    float var = sq * (1.f / 192.f) - mu * mu;
    if (tid < H2)
        hos[tid] = fmaxf((hv - mu) * rsqrtf(var + 1e-5f) * lngo[tid] + lnbo[tid], 0.f);
    __syncthreads();

    float a = bo2[tid];
    for (int i = 0; i < H2; ++i) a = fmaf(hos[i], Wo2[(size_t)i * OO + tid], a);
    out[(size_t)b * OO + tid] = a;
}

// ---------------------------------------------------------------------------
extern "C" void kernel_launch(void* const* d_in, const int* in_sizes, int n_in,
                              void* d_out, int out_size, void* d_ws, size_t ws_size,
                              hipStream_t stream) {
    const float* x    = (const float*)d_in[0];
    const float* u    = (const float*)d_in[1];
    const float* Wr1  = (const float*)d_in[2];
    const float* br1  = (const float*)d_in[3];
    const float* lng1 = (const float*)d_in[4];
    const float* lnb1 = (const float*)d_in[5];
    const float* Wr2  = (const float*)d_in[6];
    const float* br2  = (const float*)d_in[7];
    const float* lng2 = (const float*)d_in[8];
    const float* lnb2 = (const float*)d_in[9];
    const float* Wr3  = (const float*)d_in[10];
    const float* br3  = (const float*)d_in[11];
    const float* We1  = (const float*)d_in[12];
    const float* be1  = (const float*)d_in[13];
    const float* We2  = (const float*)d_in[14];
    const float* be2  = (const float*)d_in[15];
    const float* Wo1  = (const float*)d_in[16];
    const float* bo1  = (const float*)d_in[17];
    const float* lngo = (const float*)d_in[18];
    const float* lnbo = (const float*)d_in[19];
    const float* Wo2  = (const float*)d_in[20];
    const float* bo2  = (const float*)d_in[21];

    int B = in_sizes[0] / 1500;               // 2048

    float* out   = (float*)d_out;             // [B,256]
    float* rw    = out + (size_t)B * OO;      // [B,5]  routing weights (onehot)
    float* idxf  = rw + (size_t)B * NE;       // [B]    expert index
    int*   idxi  = (int*)d_ws;                // [B]    expert index (int, scratch)
    // ws layout: [0,8K) idx | [8K, 8K+1.44M) We2 B-frags | then We1 B-frags
    _Float16* Bopt2 = (_Float16*)((char*)d_ws + 8192);
    size_t bopt2_bytes = (size_t)NE * KSTEPS * NTILES * 32 * 16 * sizeof(_Float16);
    _Float16* Bopt1 = (_Float16*)((char*)d_ws + 8192 + bopt2_bytes);

    prep_bfrag2<<<NE * KSTEPS * NTILES, 32, 0, stream>>>(We2, Bopt2);
    prep_bfrag1<<<NE * NTILES, 32, 0, stream>>>(We1, Bopt1);
    router_kernel<<<B, 384, 0, stream>>>(x, u, Wr1, br1, lng1, lnb1,
                                         Wr2, br2, lng2, lnb2, Wr3, br3,
                                         rw, idxf, idxi);
    expert_kernel<<<B, 256, 0, stream>>>(x, idxi, be1, be2, Bopt1, Bopt2,
                                         Wo1, bo1, lngo, lnbo, Wo2, bo2, out);
}
